// FusionMamba_72095321030884
// MI455X (gfx1250) — matrix operand order repaired and checked by reading, not compiled, not run on hardware
//
#include <hip/hip_runtime.h>

typedef __bf16 bf16;
typedef __attribute__((ext_vector_type(8)))  __bf16 v8bf;
typedef __attribute__((ext_vector_type(16))) __bf16 v16bf;
typedef __attribute__((ext_vector_type(8)))  float  v8f;

// Problem dims
#define LB 8
#define LP 1024
#define LT 3072            // 3*LP after concat
#define DM 512
#define DI 1024
#define NST 8
#define DTR 32
#define MT (LB*LT)         // 24576 tokens
#define MP (LB*LP)         // 8192 tokens per feature

// ---------------------------------------------------------------------------
// f32 -> bf16 convert (grid-stride)
// ---------------------------------------------------------------------------
__global__ __launch_bounds__(256)
void cvt_f32_bf16(const float* __restrict__ s, bf16* __restrict__ d, long n) {
  long i = (long)blockIdx.x * 256 + threadIdx.x;
  long stride = (long)gridDim.x * 256;
  for (; i < n; i += stride) d[i] = (bf16)s[i];
}

// ---------------------------------------------------------------------------
// Shared fragment helpers (C/D per CDNA5 ISA §7.12.2; A/B k order is this code's own):
// A (16-bit, 16x32): lane holds row m=lane%16, 16 contiguous K at
//                    k+16*(lane/16) (same k order as B).
// B (32x16): lane holds column n=lane%16, 16 contiguous K at k+16*(lane/16)
//            -> contiguous 32B load from row-major weight [N,K].
// C/D: VGPR v -> row v + 8*(lane/16), col lane%16.
// ---------------------------------------------------------------------------
__device__ __forceinline__ v16bf load_afrag(const bf16* arow, int k) {
  v16bf av = *(const v16bf*)(arow + k);
  return av;
}

// ---------------------------------------------------------------------------
// Blocked bf16 WMMA GEMM: each wave computes a 16(M) x 64(N) tile of
// C[M,N] = A[M,K] * Bw[N,K]^T (+bias), 4 accumulators, A fragment reused 4x.
// Output rows remapped for the concat-along-L layout:
//   out_row = (in_row/rows_per_batch)*out_batch_stride + out_row_off
//             + in_row%rows_per_batch
// ---------------------------------------------------------------------------
template <bool OUT_BF16>
__global__ __launch_bounds__(128)
void gemm_wmma_n64(const bf16* __restrict__ A, int lda,
                   const bf16* __restrict__ Bw, int ldb,
                   void* __restrict__ Cp, int ldc,
                   const float* __restrict__ bias,
                   int M, int K,
                   int rows_per_batch, int out_batch_stride, int out_row_off) {
  const int lane = threadIdx.x & 31;
  const int wave = threadIdx.x >> 5;
  const int r    = lane & 15;
  const int half = lane >> 4;
  const int mt   = (blockIdx.y * 4 + wave) * 16;
  if (mt >= M) return;                 // wave-uniform: EXEC stays all-ones
  const int nt = blockIdx.x * 64;

  const bf16* arow = A + (size_t)(mt + r) * lda + 16 * half;
  const bf16* br0  = Bw + (size_t)(nt + r) * ldb + 16 * half;
  const bf16* br1  = br0 + (size_t)16 * ldb;
  const bf16* br2  = br0 + (size_t)32 * ldb;
  const bf16* br3  = br0 + (size_t)48 * ldb;

  v8f acc0 = {}, acc1 = {}, acc2 = {}, acc3 = {};
  for (int k = 0; k < K; k += 32) {
    v16bf av = load_afrag(arow, k);
    v16bf b0 = *(const v16bf*)(br0 + k);
    v16bf b1 = *(const v16bf*)(br1 + k);
    v16bf b2 = *(const v16bf*)(br2 + k);
    v16bf b3 = *(const v16bf*)(br3 + k);
    acc0 = __builtin_amdgcn_wmma_f32_16x16x32_bf16(false, av, false, b0,
                                                   (short)0, acc0, false, false);
    acc1 = __builtin_amdgcn_wmma_f32_16x16x32_bf16(false, av, false, b1,
                                                   (short)0, acc1, false, false);
    acc2 = __builtin_amdgcn_wmma_f32_16x16x32_bf16(false, av, false, b2,
                                                   (short)0, acc2, false, false);
    acc3 = __builtin_amdgcn_wmma_f32_16x16x32_bf16(false, av, false, b3,
                                                   (short)0, acc3, false, false);
  }

  const int batch = mt / rows_per_batch;
  const int rib   = mt - batch * rows_per_batch;
  const size_t crow0 = (size_t)batch * out_batch_stride + out_row_off + rib;
  float bv0 = 0.f, bv1 = 0.f, bv2 = 0.f, bv3 = 0.f;
  if (bias) {
    bv0 = bias[nt + r];      bv1 = bias[nt + 16 + r];
    bv2 = bias[nt + 32 + r]; bv3 = bias[nt + 48 + r];
  }
#pragma unroll
  for (int v = 0; v < 8; ++v) {
    size_t row = (crow0 + v + 8 * half) * (size_t)ldc + nt + r;
    if (OUT_BF16) {
      bf16* c = (bf16*)Cp;
      c[row]      = (bf16)(acc0[v] + bv0);
      c[row + 16] = (bf16)(acc1[v] + bv1);
      c[row + 32] = (bf16)(acc2[v] + bv2);
      c[row + 48] = (bf16)(acc3[v] + bv3);
    } else {
      float* c = (float*)Cp;
      c[row]      = acc0[v] + bv0;
      c[row + 16] = acc1[v] + bv1;
      c[row + 32] = acc2[v] + bv2;
      c[row + 48] = acc3[v] + bv3;
    }
  }
}

// ---------------------------------------------------------------------------
// Narrow variant: one 16x16 tile per wave (used for N=48 x_proj GEMM)
// ---------------------------------------------------------------------------
template <bool OUT_BF16>
__global__ __launch_bounds__(128)
void gemm_wmma(const bf16* __restrict__ A, int lda,
               const bf16* __restrict__ Bw, int ldb,
               void* __restrict__ Cp, int ldc,
               const float* __restrict__ bias,
               int M, int K,
               int rows_per_batch, int out_batch_stride, int out_row_off) {
  const int lane = threadIdx.x & 31;
  const int wave = threadIdx.x >> 5;
  const int r    = lane & 15;
  const int half = lane >> 4;
  const int mt   = (blockIdx.y * 4 + wave) * 16;
  if (mt >= M) return;
  const int nt = blockIdx.x * 16;

  const bf16* arow = A  + (size_t)(mt + r) * lda + 16 * half;
  const bf16* brow = Bw + (size_t)(nt + r) * ldb + 16 * half;

  v8f acc = {};
  for (int k = 0; k < K; k += 32) {
    v16bf av = load_afrag(arow, k);
    v16bf bv = *(const v16bf*)(brow + k);
    acc = __builtin_amdgcn_wmma_f32_16x16x32_bf16(
        false, av, false, bv, (short)0, acc, false, false);
  }

  const int batch = mt / rows_per_batch;
  const int rib   = mt - batch * rows_per_batch;
  const size_t crow0 = (size_t)batch * out_batch_stride + out_row_off + rib;
  const float bval = bias ? bias[nt + r] : 0.f;
#pragma unroll
  for (int v = 0; v < 8; ++v) {
    size_t row = crow0 + v + 8 * half;
    float val = acc[v] + bval;
    if (OUT_BF16) ((bf16*)Cp)[row * ldc + nt + r] = (bf16)val;
    else          ((float*)Cp)[row * ldc + nt + r] = val;
  }
}

// ---------------------------------------------------------------------------
// z at last token only: z[b,d] = x[b,L-1,:] . in_proj_w[DI+d,:]
// ---------------------------------------------------------------------------
__global__ __launch_bounds__(256)
void z_last(const bf16* __restrict__ xbf, const float* __restrict__ in_proj_w,
            float* __restrict__ zbuf) {
  int idx = blockIdx.x * 256 + threadIdx.x;        // 8192 = LB*DI
  int b = idx >> 10, d = idx & 1023;
  const bf16*  xr = xbf + ((size_t)b * LT + (LT - 1)) * DM;
  const float* wr = in_proj_w + (size_t)(DI + d) * DM;
  float a = 0.f;
  for (int k = 0; k < DM; ++k) a += (float)xr[k] * wr[k];
  zbuf[idx] = a;
}

// ---------------------------------------------------------------------------
// Depthwise causal conv (width 4) + SiLU.  xh/xc token-major [B*L, DI] bf16.
// ---------------------------------------------------------------------------
__global__ __launch_bounds__(256)
void dwconv_silu(const bf16* __restrict__ xh, const float* __restrict__ cw,
                 const float* __restrict__ cb, bf16* __restrict__ xc,
                 long total) {
  long idx = (long)blockIdx.x * 256 + threadIdx.x;
  if (idx >= total) return;
  int d = (int)(idx & 1023);
  long row = idx >> 10;                   // b*LT + l
  int l = (int)(row % LT);
  float acc = cb[d];
#pragma unroll
  for (int j = 0; j < 4; ++j) {
    int ll = l - 3 + j;
    if (ll >= 0) acc += (float)xh[(row - 3 + j) * DI + d] * cw[d * 4 + j];
  }
  xc[idx] = (bf16)(acc / (1.f + __expf(-acc)));   // silu
}

// ---------------------------------------------------------------------------
// Pack dt-rank slice of x_dbl (f32 [M,48]) into bf16 [M,32] for the dt GEMM
// ---------------------------------------------------------------------------
__global__ __launch_bounds__(256)
void extract_dtin(const float* __restrict__ xdbl, bf16* __restrict__ dtin,
                  long total) {
  long idx = (long)blockIdx.x * 256 + threadIdx.x;
  if (idx >= total) return;
  long row = idx >> 5;
  int c = (int)(idx & 31);
  dtin[idx] = (bf16)xdbl[row * 48 + c];
}

// ---------------------------------------------------------------------------
// Forward selective scan over L=3072, fused with:
//   - softplus(dt + dt_bias)
//   - backward-scan first step (only term surviving flip at l=L-1; A_b unused)
//   - 2*u*D skip term and silu(z) gate
// 8192 independent (b,d) lanes, 8 states each, kept in registers.
// ---------------------------------------------------------------------------
__global__ __launch_bounds__(256)
void scan_fwd(const bf16* __restrict__ dtmat, const bf16* __restrict__ ubf,
              const float* __restrict__ xdbl, const float* __restrict__ A_log,
              const float* __restrict__ dt_bias, const float* __restrict__ Dp,
              const float* __restrict__ zbuf, float* __restrict__ yfin) {
  const int b = blockIdx.x >> 2;
  const int d = ((blockIdx.x & 3) << 8) | threadIdx.x;
  float Ad[8];
#pragma unroll
  for (int n = 0; n < 8; ++n) Ad[n] = -__expf(A_log[d * 8 + n]);
  const float dtb = dt_bias[d];
  const float Dpv = Dp[d];
  float h[8] = {0, 0, 0, 0, 0, 0, 0, 0};

  const size_t base = (size_t)b * LT;
  const bf16*  dtp = dtmat + base * DI + d;
  const bf16*  up  = ubf   + base * DI + d;
  const float* xd  = xdbl  + base * 48;

  for (int l = 0; l < LT; ++l) {
    if (l + 16 < LT) {                           // global_prefetch_b8
      __builtin_prefetch(dtp + (size_t)(l + 16) * DI, 0, 1);
      __builtin_prefetch(up  + (size_t)(l + 16) * DI, 0, 1);
    }
    float x   = (float)dtp[(size_t)l * DI] + dtb;
    float dtv = (x > 20.f) ? x : __logf(1.f + __expf(x));
    float uu  = (float)up[(size_t)l * DI];
    float dtu = dtv * uu;
    const float* rp = xd + (size_t)l * 48;
    float4 B0 = *(const float4*)(rp + 32);
    float4 B1 = *(const float4*)(rp + 36);
    float Bv[8] = {B0.x, B0.y, B0.z, B0.w, B1.x, B1.y, B1.z, B1.w};
#pragma unroll
    for (int n = 0; n < 8; ++n)
      h[n] = __expf(dtv * Ad[n]) * h[n] + dtu * Bv[n];
  }

  // last-token epilogue (position L-1)
  const float* rp = xd + (size_t)(LT - 1) * 48;
  float4 B0 = *(const float4*)(rp + 32);
  float4 B1 = *(const float4*)(rp + 36);
  float4 C0 = *(const float4*)(rp + 40);
  float4 C1 = *(const float4*)(rp + 44);
  float xl   = (float)dtp[(size_t)(LT - 1) * DI] + dtb;
  float dtvl = (xl > 20.f) ? xl : __logf(1.f + __expf(xl));
  float ul   = (float)up[(size_t)(LT - 1) * DI];
  float dtul = dtvl * ul;
  float ys = h[0]*C0.x + h[1]*C0.y + h[2]*C0.z + h[3]*C0.w
           + h[4]*C1.x + h[5]*C1.y + h[6]*C1.z + h[7]*C1.w;
  float bc = B0.x*C0.x + B0.y*C0.y + B0.z*C0.z + B0.w*C0.w
           + B1.x*C1.x + B1.y*C1.y + B1.z*C1.z + B1.w*C1.w;
  float y  = ys + dtul * bc + 2.f * ul * Dpv;    // fwd + bwd-step1 + 2*u*D
  float zv = zbuf[b * DI + d];
  yfin[b * DI + d] = y * (zv / (1.f + __expf(-zv)));
}

// ---------------------------------------------------------------------------
// Fused epilogue: out_proj (last token) -> LayerNorm -> ReLU -> head
// One block per batch; LDS staging + reductions.
// ---------------------------------------------------------------------------
__global__ __launch_bounds__(256)
void head_fused(const float* __restrict__ yfin, const float* __restrict__ opw,
                const float* __restrict__ ln_g, const float* __restrict__ ln_b,
                const float* __restrict__ hw, const float* __restrict__ hb,
                float* __restrict__ out) {
  __shared__ float ysh[DI];
  __shared__ float hsh[DM];
  __shared__ float red[256];
  const int b = blockIdx.x, t = threadIdx.x;
  for (int i = t; i < DI; i += 256) ysh[i] = yfin[b * DI + i];
  __syncthreads();
  for (int e = t; e < DM; e += 256) {
    const float* w = opw + (size_t)e * DI;
    float a = 0.f;
    for (int k = 0; k < DI; ++k) a += ysh[k] * w[k];
    hsh[e] = a;
  }
  __syncthreads();
  red[t] = hsh[t] + hsh[t + 256];
  __syncthreads();
  for (int o = 128; o > 0; o >>= 1) {
    if (t < o) red[t] += red[t + o];
    __syncthreads();
  }
  float mu = red[0] / (float)DM;
  __syncthreads();
  float a0 = hsh[t] - mu, a1 = hsh[t + 256] - mu;
  red[t] = a0 * a0 + a1 * a1;
  __syncthreads();
  for (int o = 128; o > 0; o >>= 1) {
    if (t < o) red[t] += red[t + o];
    __syncthreads();
  }
  float inv = rsqrtf(red[0] / (float)DM + 1e-5f);
  __syncthreads();
  for (int e = t; e < DM; e += 256) {
    float v = (hsh[e] - mu) * inv * ln_g[e] + ln_b[e];
    hsh[e] = v > 0.f ? v : 0.f;
  }
  __syncthreads();
  if (t < 10) {
    const float* w = hw + (size_t)t * DM;
    float a = hb[t];
    for (int k = 0; k < DM; ++k) a += hsh[k] * w[k];
    out[b * 10 + t] = a;
  }
}

// ---------------------------------------------------------------------------
extern "C" void kernel_launch(void* const* d_in, const int* in_sizes, int n_in,
                              void* d_out, int out_size, void* d_ws,
                              size_t ws_size, hipStream_t stream) {
  (void)in_sizes; (void)n_in; (void)out_size; (void)ws_size;
  const float* feat0     = (const float*)d_in[0];
  const float* feat1     = (const float*)d_in[1];
  const float* feat2     = (const float*)d_in[2];
  const float* aw0       = (const float*)d_in[3];
  const float* ab0       = (const float*)d_in[4];
  const float* aw1       = (const float*)d_in[5];
  const float* ab1       = (const float*)d_in[6];
  const float* aw2       = (const float*)d_in[7];
  const float* ab2       = (const float*)d_in[8];
  const float* in_proj_w = (const float*)d_in[9];
  const float* conv_w    = (const float*)d_in[10];
  const float* conv_b    = (const float*)d_in[11];
  const float* x_proj_w  = (const float*)d_in[12];
  const float* dt_proj_w = (const float*)d_in[13];
  const float* dt_proj_b = (const float*)d_in[14];
  const float* A_log     = (const float*)d_in[15];
  /* d_in[16] = A_b_log: algebraically unused (bwd scan step 1 has h0=0) */
  const float* Dp        = (const float*)d_in[17];
  const float* out_projw = (const float*)d_in[18];
  const float* ln_g      = (const float*)d_in[19];
  const float* ln_b      = (const float*)d_in[20];
  const float* head_w    = (const float*)d_in[21];
  const float* head_b    = (const float*)d_in[22];

  char* wsp = (char*)d_ws;
  size_t off = 0;
  auto alloc = [&](size_t bytes) -> char* {
    char* p = wsp + off;
    off = (off + bytes + 511) & ~(size_t)511;
    return p;
  };
  bf16*  f0b  = (bf16*)alloc((size_t)MP * 768 * 2);
  bf16*  f1b  = (bf16*)alloc((size_t)MP * 1024 * 2);
  bf16*  f2b  = (bf16*)alloc((size_t)MP * 512 * 2);
  bf16*  aw0b = (bf16*)alloc((size_t)DM * 768 * 2);
  bf16*  aw1b = (bf16*)alloc((size_t)DM * 1024 * 2);
  bf16*  aw2b = (bf16*)alloc((size_t)DM * 512 * 2);
  bf16*  winb = (bf16*)alloc((size_t)DI * DM * 2);     // first half of in_proj
  bf16*  xpwb = (bf16*)alloc((size_t)48 * DI * 2);
  bf16*  dtwb = (bf16*)alloc((size_t)DI * DTR * 2);
  bf16*  xb   = (bf16*)alloc((size_t)MT * DM * 2);
  bf16*  xhb  = (bf16*)alloc((size_t)MT * DI * 2);
  bf16*  xcb  = (bf16*)alloc((size_t)MT * DI * 2);
  float* xdb  = (float*)alloc((size_t)MT * 48 * 4);
  bf16*  dti  = (bf16*)alloc((size_t)MT * DTR * 2);
  bf16*  dtm  = (bf16*)alloc((size_t)MT * DI * 2);
  float* zb   = (float*)alloc((size_t)LB * DI * 4);
  float* yf   = (float*)alloc((size_t)LB * DI * 4);

  // --- convert weights + features to bf16 ---
  cvt_f32_bf16<<<1024, 256, 0, stream>>>(feat0, f0b, (long)MP * 768);
  cvt_f32_bf16<<<1024, 256, 0, stream>>>(feat1, f1b, (long)MP * 1024);
  cvt_f32_bf16<<<1024, 256, 0, stream>>>(feat2, f2b, (long)MP * 512);
  cvt_f32_bf16<<<256, 256, 0, stream>>>(aw0, aw0b, (long)DM * 768);
  cvt_f32_bf16<<<256, 256, 0, stream>>>(aw1, aw1b, (long)DM * 1024);
  cvt_f32_bf16<<<256, 256, 0, stream>>>(aw2, aw2b, (long)DM * 512);
  cvt_f32_bf16<<<256, 256, 0, stream>>>(in_proj_w, winb, (long)DI * DM);
  cvt_f32_bf16<<<64, 256, 0, stream>>>(x_proj_w, xpwb, (long)48 * DI);
  cvt_f32_bf16<<<64, 256, 0, stream>>>(dt_proj_w, dtwb, (long)DI * DTR);

  // --- x = concat(f0@aw0^T, f1@aw1^T, f2@aw2^T) along L, bf16 [B*3072, 512]
  gemm_wmma_n64<true><<<dim3(DM / 64, MP / 64), 128, 0, stream>>>(
      f0b, 768, aw0b, 768, xb, DM, ab0, MP, 768, LP, LT, 0);
  gemm_wmma_n64<true><<<dim3(DM / 64, MP / 64), 128, 0, stream>>>(
      f1b, 1024, aw1b, 1024, xb, DM, ab1, MP, 1024, LP, LT, LP);
  gemm_wmma_n64<true><<<dim3(DM / 64, MP / 64), 128, 0, stream>>>(
      f2b, 512, aw2b, 512, xb, DM, ab2, MP, 512, LP, LT, 2 * LP);

  // --- z only at last token ---
  z_last<<<(LB * DI) / 256, 256, 0, stream>>>(xb, in_proj_w, zb);

  // --- xh = x @ in_proj_w[:DI]^T ---
  gemm_wmma_n64<true><<<dim3(DI / 64, MT / 64), 128, 0, stream>>>(
      xb, DM, winb, DM, xhb, DI, nullptr, MT, DM, MT, MT, 0);

  // --- depthwise conv + silu ---
  dwconv_silu<<<(unsigned)(((long)MT * DI + 255) / 256), 256, 0, stream>>>(
      xhb, conv_w, conv_b, xcb, (long)MT * DI);

  // --- x_dbl = xc @ x_proj_w^T (f32, holds dt_in | B | C), N=48 ---
  gemm_wmma<false><<<dim3(48 / 16, MT / 64), 128, 0, stream>>>(
      xcb, DI, xpwb, DI, xdb, 48, nullptr, MT, DI, MT, MT, 0);

  // --- dt = x_dbl[:, :32] @ dt_proj_w^T (K=32: single WMMA step) ---
  extract_dtin<<<(unsigned)(((long)MT * DTR + 255) / 256), 256, 0, stream>>>(
      xdb, dti, (long)MT * DTR);
  gemm_wmma_n64<true><<<dim3(DI / 64, MT / 64), 128, 0, stream>>>(
      dti, DTR, dtwb, DTR, dtm, DI, nullptr, MT, DTR, MT, MT, 0);

  // --- forward scan (+ fused bwd first step, skip term, silu(z) gate) ---
  scan_fwd<<<32, 256, 0, stream>>>(dtm, xcb, xdb, A_log, dt_proj_b, Dp, zb, yf);

  // --- out_proj(last) + LN + ReLU + head ---
  head_fused<<<LB, 256, 0, stream>>>(yf, out_projw, ln_g, ln_b, head_w, head_b,
                                     (float*)d_out);
}
